// FADiTBlockS2_88493506167001
// MI455X (gfx1250) — matrix operand order repaired and checked by reading, not compiled, not run on hardware
//
#include <hip/hip_runtime.h>

// ---------------- types ----------------
typedef __attribute__((ext_vector_type(16))) __bf16 v16bf;
typedef __attribute__((ext_vector_type(8)))  float  v8f;

union ABu { v16bf v; uint4 q[2]; };

__device__ inline float bf2f(unsigned short h){
  union { unsigned u; float f; } v; v.u = ((unsigned)h) << 16; return v.f;
}
__device__ inline unsigned short f2bf(float f){
  union { float f; unsigned u; } v; v.f = f;
  unsigned r = v.u + 0x7FFFu + ((v.u >> 16) & 1u);
  return (unsigned short)(r >> 16);
}
__device__ inline float gelu_f(float x){
  float x3 = x * x * x;
  return 0.5f * x * (1.0f + tanhf(0.7978845608028654f * (x + 0.044715f * x3)));
}

// ---------------- generic batched bf16 WMMA GEMM ----------------
// C[z,m,n] = sum_k A[z,m,k]*B[z,n,k]   (B supplied "transposed": k-stride 1)
// index decompositions: m=(m1,m0) via md0s, n=(n1,n0) via nd0s, z=(z2,z1,z0).
struct GemmP {
  const unsigned short* A;
  const unsigned short* B;
  float* Cf;                 // optional f32 output
  unsigned short* Ch;        // optional bf16 output
  const float* bias;         // optional bias[n]
  const float* gate;         // optional: out = resid + gate[z,n]*val
  const float* resid;        // same strides as C
  long long gate_sb;
  int M, N, K;
  int md0s, nd0s, zd0s, zd1s;
  int gelu_flag;
  long long Asb2, Asb1, Asb0, Asm1, Asm0;
  long long Bsb2, Bsb1, Bsb0, Bsn1, Bsn0;
  long long Csb2, Csb1, Csb0, Csm1, Csm0, Csn1, Csn0;
};

__device__ inline void epilogue_tile(const GemmP& p, v8f acc,
                                     long long m_base, long long n_base,
                                     int ti, int tj, int lane, int z, long long cz,
                                     long long mmask, long long nmask)
{
  long long mr = m_base + ti * 16 + ((lane & 16) ? 8 : 0);
  long long nc = n_base + tj * 16 + (lane & 15);
  long long cn = cz + (nc >> p.nd0s) * p.Csn1 + (nc & nmask) * p.Csn0;
  float bz = p.bias ? p.bias[nc] : 0.0f;
  float gt = p.gate ? p.gate[(long long)z * p.gate_sb + nc] : 0.0f;
#pragma unroll
  for (int vv = 0; vv < 8; ++vv){
    long long mm = mr + vv;
    long long o = cn + (mm >> p.md0s) * p.Csm1 + (mm & mmask) * p.Csm0;
    float val = acc[vv] + bz;
    if (p.gelu_flag) val = gelu_f(val);
    if (p.gate)      val = p.resid[o] + gt * val;
    if (p.Cf) p.Cf[o] = val;
    if (p.Ch) p.Ch[o] = f2bf(val);
  }
}

__global__ __launch_bounds__(256) void gemm_bf16_wmma(GemmP p)
{
  const int lane = threadIdx.x & 31;
  const int wave = threadIdx.x >> 5;
  const long long m_base = (long long)blockIdx.y * 128 + (wave >> 1) * 32;
  const long long n_base = (long long)blockIdx.x * 64  + (wave & 1)  * 32;
  const int z  = blockIdx.z;
  const int z0 = z & ((1 << p.zd0s) - 1);
  const int z1 = (z >> p.zd0s) & ((1 << p.zd1s) - 1);
  const int z2 = z >> (p.zd0s + p.zd1s);
  const long long mmask = (1ll << p.md0s) - 1;
  const long long nmask = (1ll << p.nd0s) - 1;

  const unsigned short* A  = p.A + (long long)z2 * p.Asb2 + (long long)z1 * p.Asb1 + (long long)z0 * p.Asb0;
  const unsigned short* Bp = p.B + (long long)z2 * p.Bsb2 + (long long)z1 * p.Bsb1 + (long long)z0 * p.Bsb0;
  const long long cz = (long long)z2 * p.Csb2 + (long long)z1 * p.Csb1 + (long long)z0 * p.Csb0;

  // A (16xK) lane layout: lane<16 -> elems {k..k+7, k+16..k+23}; lane>=16 -> +8
  const int r     = lane & 15;
  const int koffA = (lane & 16) ? 8  : 0;
  const int koffB = (lane & 16) ? 8  : 0;
  const long long mA0 = m_base + r, mA1 = m_base + 16 + r;
  const unsigned short* Ar0 = A + (mA0 >> p.md0s) * p.Asm1 + (mA0 & mmask) * p.Asm0;
  const unsigned short* Ar1 = A + (mA1 >> p.md0s) * p.Asm1 + (mA1 & mmask) * p.Asm0;
  const long long nB0 = n_base + r, nB1 = n_base + 16 + r;
  const unsigned short* Bc0 = Bp + (nB0 >> p.nd0s) * p.Bsn1 + (nB0 & nmask) * p.Bsn0;
  const unsigned short* Bc1 = Bp + (nB1 >> p.nd0s) * p.Bsn1 + (nB1 & nmask) * p.Bsn0;

  v8f acc00 = {0,0,0,0,0,0,0,0}, acc01 = {0,0,0,0,0,0,0,0};
  v8f acc10 = {0,0,0,0,0,0,0,0}, acc11 = {0,0,0,0,0,0,0,0};

  for (int kk = 0; kk < p.K; kk += 32){
    ABu a0, a1, b0, b1;
    a0.q[0] = *(const uint4*)(Ar0 + kk + koffA);
    a0.q[1] = *(const uint4*)(Ar0 + kk + 16 + koffA);
    a1.q[0] = *(const uint4*)(Ar1 + kk + koffA);
    a1.q[1] = *(const uint4*)(Ar1 + kk + 16 + koffA);
    b0.q[0] = *(const uint4*)(Bc0 + kk + koffB);
    b0.q[1] = *(const uint4*)(Bc0 + kk + 16 + koffB);
    b1.q[0] = *(const uint4*)(Bc1 + kk + koffB);
    b1.q[1] = *(const uint4*)(Bc1 + kk + 16 + koffB);
    acc00 = __builtin_amdgcn_wmma_f32_16x16x32_bf16(false, a0.v, false, b0.v, (short)0, acc00, false, false);
    acc01 = __builtin_amdgcn_wmma_f32_16x16x32_bf16(false, a0.v, false, b1.v, (short)0, acc01, false, false);
    acc10 = __builtin_amdgcn_wmma_f32_16x16x32_bf16(false, a1.v, false, b0.v, (short)0, acc10, false, false);
    acc11 = __builtin_amdgcn_wmma_f32_16x16x32_bf16(false, a1.v, false, b1.v, (short)0, acc11, false, false);
  }

  epilogue_tile(p, acc00, m_base, n_base, 0, 0, lane, z, cz, mmask, nmask);
  epilogue_tile(p, acc01, m_base, n_base, 0, 1, lane, z, cz, mmask, nmask);
  epilogue_tile(p, acc10, m_base, n_base, 1, 0, lane, z, cz, mmask, nmask);
  epilogue_tile(p, acc11, m_base, n_base, 1, 1, lane, z, cz, mmask, nmask);
}

// ---------------- small support kernels ----------------

// weight transpose+convert: W[K,N] f32 -> Wt[N,K] bf16
__global__ void wt_kernel(const float* __restrict__ W, unsigned short* __restrict__ Wt, int K, int N){
  long long id  = (long long)blockIdx.x * 256 + threadIdx.x;
  long long tot = (long long)K * N;
  if (id >= tot) return;
  int k = (int)(id % K);
  long long n = id / K;
  Wt[id] = f2bf(W[(long long)k * N + n]);
}

// mod[b,j] = cond[b,:] @ adaLN_w[:,j] + adaLN_b[j]   (4 x 1536)
__global__ void mod_kernel(const float* __restrict__ cond, const float* __restrict__ W,
                           const float* __restrict__ bias, float* __restrict__ modp){
  int id = blockIdx.x * 256 + threadIdx.x;   // 6144 total
  int b = id / 1536, j = id % 1536;
  float acc = bias[j];
  for (int c = 0; c < 256; ++c) acc += cond[b * 256 + c] * W[c * 1536 + j];
  modp[id] = acc;
}

// lwn[i] = cos(lat[i]) / mean(cos) / nlat
__global__ void lw_kernel(const float* __restrict__ lat, float* __restrict__ lwn){
  __shared__ float sb[128];
  int i = threadIdx.x;
  float c = __cosf(lat[i]);
  sb[i] = c; __syncthreads();
  for (int s = 64; s; s >>= 1){ if (i < s) sb[i] += sb[i + s]; __syncthreads(); }
  float mean = sb[0] * (1.0f / 128.0f);
  lwn[i] = c / mean * (1.0f / 128.0f);
}

// per-token LayerNorm with adaLN shift/scale -> bf16. 1 wave per token, 8 waves/block.
__global__ __launch_bounds__(256) void ln_mod_kernel(const float* __restrict__ x,
    const float* __restrict__ modp, unsigned short* __restrict__ outb,
    int sh_off, int sc_off)
{
  const int lane = threadIdx.x & 31;
  const int w    = threadIdx.x >> 5;
  const long long t = (long long)blockIdx.x * 8 + w;
  const float* xr = x + t * 256;
  const int b = (int)(t >> 15);              // 32768 tokens per batch entry
  float4 v0 = ((const float4*)xr)[lane * 2];
  float4 v1 = ((const float4*)xr)[lane * 2 + 1];
  float s = v0.x + v0.y + v0.z + v0.w + v1.x + v1.y + v1.z + v1.w;
  float q = v0.x*v0.x + v0.y*v0.y + v0.z*v0.z + v0.w*v0.w
          + v1.x*v1.x + v1.y*v1.y + v1.z*v1.z + v1.w*v1.w;
  for (int m = 16; m; m >>= 1){ s += __shfl_xor(s, m, 32); q += __shfl_xor(q, m, 32); }
  float mean = s * (1.0f / 256.0f);
  float var  = q * (1.0f / 256.0f) - mean * mean;
  float rs   = rsqrtf(var + 1e-5f);
  const float* mb = modp + (long long)b * 1536;
  float xv[8] = {v0.x, v0.y, v0.z, v0.w, v1.x, v1.y, v1.z, v1.w};
  unsigned short o[8];
#pragma unroll
  for (int e = 0; e < 8; ++e){
    int c = lane * 8 + e;
    float y = (xv[e] - mean) * rs;
    o[e] = f2bf(y * (1.0f + mb[sc_off + c]) + mb[sh_off + c]);
  }
  uint4 pack;
  pack.x = (unsigned)o[0] | ((unsigned)o[1] << 16);
  pack.y = (unsigned)o[2] | ((unsigned)o[3] << 16);
  pack.z = (unsigned)o[4] | ((unsigned)o[5] << 16);
  pack.w = (unsigned)o[6] | ((unsigned)o[7] << 16);
  *(uint4*)(outb + t * 256 + lane * 8) = pack;
}

// t_x[b,l,c] = sum_i um[b,i,l,c]*lwn[i]   (lwn carries /nlat and cos-normalization)
__global__ __launch_bounds__(256) void tx_kernel(const unsigned short* __restrict__ um,
    const float* __restrict__ lwn, unsigned short* __restrict__ txb)
{
  const int c  = threadIdx.x;
  const int bl = blockIdx.x;        // b*256 + l
  const int b  = bl >> 8, l = bl & 255;
  float acc = 0.f;
  for (int i = 0; i < 128; ++i)
    acc += bf2f(um[((((long long)b * 128 + i) * 256 + l) * 256) + c]) * lwn[i];
  txb[(long long)bl * 256 + c] = f2bf(acc);
}

// t_y[b,i,c] = mean_l um[b,i,l,c]
__global__ __launch_bounds__(256) void ty_kernel(const unsigned short* __restrict__ um,
                                                 unsigned short* __restrict__ tyb)
{
  const int c = threadIdx.x;
  const long long bi = blockIdx.x;  // b*128 + i
  float acc = 0.f;
  for (int l = 0; l < 256; ++l) acc += bf2f(um[(bi * 256 + l) * 256 + c]);
  tyb[bi * 256 + c] = f2bf(acc * (1.0f / 256.0f));
}

// radial: out[h, id] = sum_n basis(d[id], n) * w[n-1, h]
__global__ void radial_kernel(const float* __restrict__ d, const float* __restrict__ w,
                              float* __restrict__ out, int n2)
{
  int id = blockIdx.x * 256 + threadIdx.x;
  if (id >= n2) return;
  float dv = d[id];
  float acc[8];
#pragma unroll
  for (int h = 0; h < 8; ++h) acc[h] = 0.f;
  for (int n = 1; n <= 32; ++n){
    float basis = (dv > 1e-6f) ? (__sinf(dv * (float)n) / dv) : (float)n;
    const float* wr = w + (n - 1) * 8;
#pragma unroll
    for (int h = 0; h < 8; ++h) acc[h] += basis * wr[h];
  }
  for (int h = 0; h < 8; ++h) out[(long long)h * n2 + id] = acc[h];
}

// row softmax of S*R -> bf16. 1 wave per row. hmask extracts (h*Lr+i) from row id.
__global__ __launch_bounds__(256) void softmax_kernel(const float* __restrict__ S,
    const float* __restrict__ R, unsigned short* __restrict__ outb, int L, int hmask)
{
  const int lane = threadIdx.x & 31;
  const int w    = threadIdx.x >> 5;
  const long long row = (long long)blockIdx.x * 8 + w;
  const float* sp = S + row * L;
  const float* rp = R + (long long)(row & hmask) * L;
  const int epl = L >> 5;
  float vals[8];
  float mx = -1e30f;
  for (int e = 0; e < epl; ++e){
    int j = e * 32 + lane;
    float xv = sp[j] * rp[j];
    vals[e] = xv; mx = fmaxf(mx, xv);
  }
  for (int m = 16; m; m >>= 1) mx = fmaxf(mx, __shfl_xor(mx, m, 32));
  float sum = 0.f;
  for (int e = 0; e < epl; ++e){ vals[e] = __expf(vals[e] - mx); sum += vals[e]; }
  for (int m = 16; m; m >>= 1) sum += __shfl_xor(sum, m, 32);
  float inv = 1.0f / sum;
  for (int e = 0; e < epl; ++e) outb[row * L + e * 32 + lane] = f2bf(vals[e] * inv);
}

// GroupNorm over 64-wide head chunks; block = 1 token (8 waves = 8 heads)
__global__ __launch_bounds__(256) void gn_kernel(const float* __restrict__ up,
                                                 unsigned short* __restrict__ outb)
{
  const int lane = threadIdx.x & 31;
  const int h    = threadIdx.x >> 5;
  const long long t = blockIdx.x;
  const float* pp = up + t * 512 + h * 64;
  float2 v = ((const float2*)pp)[lane];
  float s = v.x + v.y;
  float q = v.x * v.x + v.y * v.y;
  for (int m = 16; m; m >>= 1){ s += __shfl_xor(s, m, 32); q += __shfl_xor(q, m, 32); }
  float mean = s * (1.0f / 64.0f);
  float var  = q * (1.0f / 64.0f) - mean * mean;
  float rs   = rsqrtf(var + 1e-6f);
  unsigned short a  = f2bf((v.x - mean) * rs);
  unsigned short bq = f2bf((v.y - mean) * rs);
  *(unsigned*)(outb + t * 512 + h * 64 + lane * 2) = (unsigned)a | ((unsigned)bq << 16);
}

// ---------------- host ----------------
static int lg2i(int x){ int s = 0; while ((1 << s) < x) ++s; return s; }

extern "C" void kernel_launch(void* const* d_in, const int* in_sizes, int n_in,
                              void* d_out, int out_size, void* d_ws, size_t ws_size,
                              hipStream_t stream)
{
  (void)in_sizes; (void)n_in; (void)out_size; (void)ws_size;
  const float* u    = (const float*)d_in[0];
  const float* lat  = (const float*)d_in[1];
  const float* latd = (const float*)d_in[2];
  const float* lond = (const float*)d_in[3];
  const float* cond = (const float*)d_in[4];
  const float* adaW = (const float*)d_in[5];
  const float* adaB = (const float*)d_in[6];
  const float* w_v  = (const float*)d_in[7];
  const float* w_xin= (const float*)d_in[8];
  const float* w_xw1= (const float*)d_in[9];
  const float* b_x1 = (const float*)d_in[10];
  const float* w_xw2= (const float*)d_in[11];
  const float* b_x2 = (const float*)d_in[12];
  const float* w_yin= (const float*)d_in[13];
  const float* w_yw1= (const float*)d_in[14];
  const float* b_y1 = (const float*)d_in[15];
  const float* w_yw2= (const float*)d_in[16];
  const float* b_y2 = (const float*)d_in[17];
  const float* w_kx = (const float*)d_in[18];
  const float* w_ky = (const float*)d_in[19];
  const float* rxw  = (const float*)d_in[20];
  const float* ryw  = (const float*)d_in[21];
  const float* w_mg = (const float*)d_in[22];
  const float* b_mg = (const float*)d_in[23];
  const float* w_f1 = (const float*)d_in[24];
  const float* b_f1 = (const float*)d_in[25];
  const float* w_f2 = (const float*)d_in[26];
  const float* b_f2 = (const float*)d_in[27];
  float* outp = (float*)d_out;

  const long long T = 131072; // 4*128*256 tokens

  char* ws = (char*)d_ws;
  size_t off = 0;
  auto alloc = [&](size_t bytes)->char*{
    size_t o = off; off += (bytes + 255) & ~(size_t)255; return ws + o;
  };

  float* modp = (float*)alloc(4 * 1536 * 4);
  float* lwn  = (float*)alloc(128 * 4);
  unsigned short* um_bf  = (unsigned short*)alloc((size_t)T * 256 * 2);
  unsigned short* wt_v   = (unsigned short*)alloc(512 * 256 * 2);
  unsigned short* wt_xin = (unsigned short*)alloc(256 * 256 * 2);
  unsigned short* wt_xw1 = (unsigned short*)alloc(256 * 256 * 2);
  unsigned short* wt_xw2 = (unsigned short*)alloc(128 * 256 * 2);
  unsigned short* wt_yin = (unsigned short*)alloc(256 * 256 * 2);
  unsigned short* wt_yw1 = (unsigned short*)alloc(256 * 256 * 2);
  unsigned short* wt_yw2 = (unsigned short*)alloc(128 * 256 * 2);
  unsigned short* wt_kx  = (unsigned short*)alloc(1024 * 128 * 2);
  unsigned short* wt_ky  = (unsigned short*)alloc(1024 * 128 * 2);
  unsigned short* wt_mg  = (unsigned short*)alloc(256 * 512 * 2);
  unsigned short* wt_f1  = (unsigned short*)alloc(1024 * 256 * 2);
  unsigned short* wt_f2  = (unsigned short*)alloc(256 * 1024 * 2);
  unsigned short* txb    = (unsigned short*)alloc(1024 * 256 * 2);
  unsigned short* tyb    = (unsigned short*)alloc(512 * 256 * 2);
  unsigned short* xa_bf  = (unsigned short*)alloc(1024 * 256 * 2);
  unsigned short* xh_bf  = (unsigned short*)alloc(1024 * 256 * 2);
  unsigned short* ux_bf  = (unsigned short*)alloc(1024 * 128 * 2);
  unsigned short* ya_bf  = (unsigned short*)alloc(512 * 256 * 2);
  unsigned short* yh_bf  = (unsigned short*)alloc(512 * 256 * 2);
  unsigned short* uy_bf  = (unsigned short*)alloc(512 * 128 * 2);
  unsigned short* qkx_bf = (unsigned short*)alloc((size_t)4 * 2 * 8 * 256 * 64 * 2);
  unsigned short* qky_bf = (unsigned short*)alloc((size_t)4 * 2 * 8 * 128 * 64 * 2);
  float* Sx = (float*)alloc((size_t)4 * 8 * 256 * 256 * 4);
  float* Sy = (float*)alloc((size_t)4 * 8 * 128 * 128 * 4);
  float* rx = (float*)alloc((size_t)8 * 256 * 256 * 4);
  float* ry = (float*)alloc((size_t)8 * 128 * 128 * 4);
  unsigned short* kx_bf  = (unsigned short*)alloc((size_t)4 * 8 * 256 * 256 * 2);
  unsigned short* ky_bf  = (unsigned short*)alloc((size_t)4 * 8 * 128 * 128 * 2);
  unsigned short* v_t    = (unsigned short*)alloc((size_t)4 * 8 * 256 * 64 * 128 * 2);
  unsigned short* phi1_t = (unsigned short*)alloc((size_t)4 * 8 * 128 * 64 * 256 * 2);
  float* u_phi = (float*)alloc((size_t)T * 512 * 4);
  unsigned short* gn_bf = (unsigned short*)alloc((size_t)T * 512 * 2);
  float* u1 = (float*)alloc((size_t)T * 256 * 4);
  unsigned short* u2_bf = (unsigned short*)alloc((size_t)T * 256 * 2);
  unsigned short* h1_bf = (unsigned short*)alloc((size_t)T * 1024 * 2);

  auto run = [&](GemmP& p, int Z){
    dim3 g(p.N / 64, p.M / 128, Z);
    gemm_bf16_wmma<<<g, 256, 0, stream>>>(p);
  };
  auto simple = [&](const unsigned short* A, const unsigned short* Bw,
                    unsigned short* Ch, const float* bias, int gel,
                    int M, int N, int K){
    GemmP p{}; p.A = A; p.B = Bw; p.Ch = Ch; p.bias = bias; p.gelu_flag = gel;
    p.M = M; p.N = N; p.K = K;
    p.md0s = lg2i(M); p.nd0s = lg2i(N);
    p.Asm0 = K; p.Bsn0 = K; p.Csm0 = N; p.Csn0 = 1;
    run(p, 1);
  };
  auto wt = [&](const float* W, unsigned short* Wt, int K, int N){
    int tot = K * N;
    wt_kernel<<<(tot + 255) / 256, 256, 0, stream>>>(W, Wt, K, N);
  };

  // 1) weights -> bf16 transposed [N][K]
  wt(w_v,   wt_v,   256, 512);
  wt(w_xin, wt_xin, 256, 256);
  wt(w_xw1, wt_xw1, 256, 256);
  wt(w_xw2, wt_xw2, 256, 128);
  wt(w_yin, wt_yin, 256, 256);
  wt(w_yw1, wt_yw1, 256, 256);
  wt(w_yw2, wt_yw2, 256, 128);
  wt(w_kx,  wt_kx,  128, 1024);
  wt(w_ky,  wt_ky,  128, 1024);
  wt(w_mg,  wt_mg,  512, 256);
  wt(w_f1,  wt_f1,  256, 1024);
  wt(w_f2,  wt_f2,  1024, 256);

  // 2) adaLN modulation + lat weights
  mod_kernel<<<24, 256, 0, stream>>>(cond, adaW, adaB, modp);
  lw_kernel<<<1, 128, 0, stream>>>(lat, lwn);

  // 3) um = LN(u)*(1+sc_msa)+sh_msa  -> bf16
  ln_mod_kernel<<<16384, 256, 0, stream>>>(u, modp, um_bf, /*sh*/0, /*sc*/256);

  // 4) lat/lon reductions
  tx_kernel<<<1024, 256, 0, stream>>>(um_bf, lwn, txb);
  ty_kernel<<<512, 256, 0, stream>>>(um_bf, tyb);

  // 5) bottleneck MLPs (u_x, u_y)
  simple(txb,   wt_xin, xa_bf, nullptr, 0, 1024, 256, 256);
  simple(xa_bf, wt_xw1, xh_bf, b_x1,   1, 1024, 256, 256);
  simple(xh_bf, wt_xw2, ux_bf, b_x2,   0, 1024, 128, 256);
  simple(tyb,   wt_yin, ya_bf, nullptr, 0, 512, 256, 256);
  simple(ya_bf, wt_yw1, yh_bf, b_y1,   1, 512, 256, 256);
  simple(yh_bf, wt_yw2, uy_bf, b_y2,   0, 512, 128, 256);

  // 6) q/k projections -> layout [b][s][h][row][c]
  { GemmP p{}; p.A = ux_bf; p.B = wt_kx; p.Ch = qkx_bf;
    p.M = 1024; p.N = 1024; p.K = 128; p.md0s = 8; p.nd0s = 6;
    p.Asm1 = 32768; p.Asm0 = 128; p.Bsn1 = 8192; p.Bsn0 = 128;
    p.Csm1 = 262144; p.Csm0 = 64; p.Csn1 = 16384; p.Csn0 = 1; run(p, 1); }
  { GemmP p{}; p.A = uy_bf; p.B = wt_ky; p.Ch = qky_bf;
    p.M = 512; p.N = 1024; p.K = 128; p.md0s = 7; p.nd0s = 6;
    p.Asm1 = 16384; p.Asm0 = 128; p.Bsn1 = 8192; p.Bsn0 = 128;
    p.Csm1 = 131072; p.Csm0 = 64; p.Csn1 = 8192; p.Csn0 = 1; run(p, 1); }

  // 7) attention scores S = q @ k^T per (b,h)
  { GemmP p{}; p.A = qkx_bf; p.B = qkx_bf + 131072; p.Cf = Sx;
    p.M = 256; p.N = 256; p.K = 64; p.md0s = 8; p.nd0s = 8; p.zd0s = 3;
    p.Asb2 = 262144; p.Asb0 = 16384; p.Asm0 = 64;
    p.Bsb2 = 262144; p.Bsb0 = 16384; p.Bsn0 = 64;
    p.Csb2 = 524288; p.Csb0 = 65536; p.Csm0 = 256; p.Csn0 = 1; run(p, 32); }
  { GemmP p{}; p.A = qky_bf; p.B = qky_bf + 65536; p.Cf = Sy;
    p.M = 128; p.N = 128; p.K = 64; p.md0s = 7; p.nd0s = 7; p.zd0s = 3;
    p.Asb2 = 131072; p.Asb0 = 8192; p.Asm0 = 64;
    p.Bsb2 = 131072; p.Bsb0 = 8192; p.Bsn0 = 64;
    p.Csb2 = 131072; p.Csb0 = 16384; p.Csm0 = 128; p.Csn0 = 1; run(p, 32); }

  // 8) radial modulation + softmax -> bf16 attention matrices
  radial_kernel<<<256, 256, 0, stream>>>(lond, rxw, rx, 65536);
  radial_kernel<<<64, 256, 0, stream>>>(latd, ryw, ry, 16384);
  softmax_kernel<<<1024, 256, 0, stream>>>(Sx, rx, kx_bf, 256, 2047);
  softmax_kernel<<<512, 256, 0, stream>>>(Sy, ry, ky_bf, 128, 1023);

  // 9) V projection -> v_t[b][h][lon][c][lat]  (k-stride-1 for next GEMM)
  { GemmP p{}; p.A = um_bf; p.B = wt_v; p.Ch = v_t;
    p.M = 32768; p.N = 64; p.K = 256; p.md0s = 8; p.nd0s = 6; p.zd0s = 3;
    p.Asb2 = 8388608; p.Asb0 = 0; p.Asm1 = 65536; p.Asm0 = 256;
    p.Bsb0 = 16384; p.Bsn0 = 256;
    p.Csb2 = 16777216; p.Csb0 = 2097152; p.Csm1 = 1; p.Csm0 = 8192; p.Csn0 = 128;
    run(p, 32); }

  // 10) phi1 = k_y @ V -> phi1_t[b][h][lat][c][lon]
  { GemmP p{}; p.A = ky_bf; p.B = v_t; p.Ch = phi1_t;
    p.M = 128; p.N = 16384; p.K = 128; p.md0s = 7; p.nd0s = 6; p.zd0s = 3;
    p.Asb2 = 131072; p.Asb0 = 16384; p.Asm0 = 128;
    p.Bsb2 = 16777216; p.Bsb0 = 2097152; p.Bsn1 = 8192; p.Bsn0 = 128;
    p.Csb2 = 16777216; p.Csb0 = 2097152; p.Csm0 = 16384; p.Csn1 = 1; p.Csn0 = 256;
    run(p, 32); }

  // 11) u_phi = k_x @ phi1 -> (b, lat, lon, h*64+c) f32
  { GemmP p{}; p.A = kx_bf; p.B = phi1_t; p.Cf = u_phi;
    p.M = 256; p.N = 64; p.K = 256; p.md0s = 8; p.nd0s = 6; p.zd0s = 7; p.zd1s = 3;
    p.Asb2 = 524288; p.Asb1 = 65536; p.Asb0 = 0; p.Asm0 = 256;
    p.Bsb2 = 16777216; p.Bsb1 = 2097152; p.Bsb0 = 16384; p.Bsn0 = 256;
    p.Csb2 = 16777216; p.Csb1 = 64; p.Csb0 = 131072; p.Csm0 = 512; p.Csn0 = 1;
    run(p, 4096); }

  // 12) per-head GroupNorm -> bf16
  gn_kernel<<<131072, 256, 0, stream>>>(u_phi, gn_bf);

  // 13) merge + residual gate: u1 = u + g_msa * (gn@merge_w + b)
  { GemmP p{}; p.A = gn_bf; p.B = wt_mg; p.Cf = u1;
    p.bias = b_mg; p.gate = modp + 512; p.resid = u; p.gate_sb = 1536;
    p.M = 32768; p.N = 256; p.K = 512; p.md0s = 15; p.nd0s = 8; p.zd0s = 2;
    p.Asb0 = 16777216; p.Asm0 = 512; p.Bsn0 = 512;
    p.Csb0 = 8388608; p.Csm0 = 256; p.Csn0 = 1; run(p, 4); }

  // 14) u2 = LN(u1)*(1+sc_mlp)+sh_mlp -> bf16
  ln_mod_kernel<<<16384, 256, 0, stream>>>(u1, modp, u2_bf, /*sh*/768, /*sc*/1024);

  // 15) FFN
  { GemmP p{}; p.A = u2_bf; p.B = wt_f1; p.Ch = h1_bf;
    p.bias = b_f1; p.gelu_flag = 1;
    p.M = 32768; p.N = 1024; p.K = 256; p.md0s = 15; p.nd0s = 10; p.zd0s = 2;
    p.Asb0 = 8388608; p.Asm0 = 256; p.Bsn0 = 256;
    p.Csb0 = 33554432; p.Csm0 = 1024; p.Csn0 = 1; run(p, 4); }
  { GemmP p{}; p.A = h1_bf; p.B = wt_f2; p.Cf = outp;
    p.bias = b_f2; p.gate = modp + 1280; p.resid = u1; p.gate_sb = 1536;
    p.M = 32768; p.N = 256; p.K = 1024; p.md0s = 15; p.nd0s = 8; p.zd0s = 2;
    p.Asb0 = 33554432; p.Asm0 = 1024; p.Bsn0 = 1024;
    p.Csb0 = 8388608; p.Csm0 = 256; p.Csn0 = 1; run(p, 4); }
}
